// Decoder_4604204941441
// MI455X (gfx1250) — compile-verified, matrix operand order checked by reading, not run on hardware
//
#include <hip/hip_runtime.h>
#include <hip/hip_bf16.h>
#include <math.h>

// Problem dims (fixed by reference)
#define B_ 128
#define L_ 128
#define E_ 512
#define H_ 512

typedef __attribute__((ext_vector_type(16))) __bf16 v16bf;
typedef __attribute__((ext_vector_type(8)))  float  v8f;

// 16-byte int vector type matching the async-to-LDS builtin's pointee type
typedef int async_b128_t __attribute__((vector_size(16)));
typedef __attribute__((address_space(1))) async_b128_t* async_gptr;
typedef __attribute__((address_space(3))) async_b128_t* async_lptr;

// ---------- helpers ----------
__device__ __forceinline__ unsigned short f32_to_bf16_rne(float f) {
  unsigned int u = __float_as_uint(f);
  u += 0x7FFFu + ((u >> 16) & 1u);           // round-to-nearest-even
  return (unsigned short)(u >> 16);
}
__device__ __forceinline__ float sigmoidf_(float x) {
  return 1.0f / (1.0f + __expf(-x));
}

// ---------- elementwise conversion / init ----------
__global__ void convert_f32_bf16(const float* __restrict__ src,
                                 unsigned short* __restrict__ dst, int n) {
  int i = blockIdx.x * blockDim.x + threadIdx.x;
  if (i < n) dst[i] = f32_to_bf16_rne(src[i]);
}

__global__ void init_state(const float* __restrict__ c0,
                           float* __restrict__ c, float* __restrict__ mask) {
  int i = blockIdx.x * blockDim.x + threadIdx.x;
  if (i < B_ * H_) c[i] = c0[i];
  if (i < B_ * L_) mask[i] = 1.0f;
}

// ---------- WMMA GEMM with async-LDS-staged weight panel ----------
// D[M,N] = act(A[M,K](bf16) @ W[N,K]^T + bias).
// One block = 4 waves stacked in M sharing one 16-column weight panel (nt).
// The 16 x KDIM bf16 panel is staged LDS-side with GLOBAL_LOAD_ASYNC_TO_LDS_B128
// (ASYNCcnt), rows padded by 8 elements so the 16 row-readers are bank-conflict
// free; fragments then come from ds_load_b128 while A streams from L2.
template <int ACT, int KDIM> // ACT: 0 = identity, 1 = tanh
__global__ void __launch_bounds__(128)
wmma_gemm_lds(const unsigned short* __restrict__ A, int lda,
              const unsigned short* __restrict__ W,   // [N, KDIM] row-major
              const float* __restrict__ bias,
              int ntilesN,
              unsigned short* __restrict__ Dbf, int lddb,
              float* __restrict__ Df, int lddf) {
  constexpr int KP = KDIM + 8;                       // padded LDS row (elements)
  __shared__ unsigned short bpanel[16 * KP];

  const int nt  = blockIdx.x % ntilesN;
  const int mtg = blockIdx.x / ntilesN;
  const int tid = threadIdx.x;

  // ---- stage the 16 x KDIM weight panel into LDS ----
  const int chunks = 16 * (KDIM / 8);                // 16-byte transfers
  for (int idx = tid; idx < chunks; idx += 128) {
    const int r = idx / (KDIM / 8);
    const int c = idx % (KDIM / 8);
    const unsigned short* g = W + (size_t)(nt * 16 + r) * KDIM + c * 8;
    unsigned short*       l = &bpanel[r * KP + c * 8];
#if __has_builtin(__builtin_amdgcn_global_load_async_to_lds_b128)
    __builtin_amdgcn_global_load_async_to_lds_b128(
        (async_gptr)g, (async_lptr)l, 0, 0);
#else
    *(int4*)l = *(const int4*)g;                     // synchronous fallback
#endif
  }
#if __has_builtin(__builtin_amdgcn_global_load_async_to_lds_b128)
  asm volatile("s_wait_asynccnt 0x0" ::: "memory");  // our async writes done
#endif
  __syncthreads();                                   // panel visible to all waves

  // ---- per-wave 16x16 tile ----
  const int lane  = tid & 31;
  const int row   = lane & 15;
  const int khalf = (lane >> 4) << 4;
  const int mt    = mtg * 4 + (tid >> 5);

  const unsigned short* aRow = A + (size_t)(mt * 16 + row) * lda + khalf;
  const int             bOff = row * KP + khalf;

  v8f acc = {};
  for (int k = 0; k < KDIM; k += 32) {
    v16bf a = *reinterpret_cast<const v16bf*>(aRow + k);
    v16bf b = *reinterpret_cast<const v16bf*>(&bpanel[bOff + k]);
    acc = __builtin_amdgcn_wmma_f32_16x16x32_bf16(false, a, false, b,
                                                  (short)0, acc, false, false);
  }

  const int   n     = nt * 16 + (lane & 15);
  const float bv    = bias ? bias[n] : 0.0f;
  const int   mbase = mt * 16 + ((lane >> 4) << 3);  // lanes 16-31 hold M = 8..15
#pragma unroll
  for (int r = 0; r < 8; ++r) {
    float v = acc[r] + bv;
    if (ACT == 1) v = tanhf(v);
    const int m = mbase + r;
    if (Dbf) Dbf[(size_t)m * lddb + n] = f32_to_bf16_rne(v);
    if (Df)  Df[(size_t)m * lddf + n]  = v;
  }
}

// ---------- fused LSTM gates GEMM + cell ----------
// Wave computes a 16x16 tile of hidden columns j, for all four gate chunks
// (rows j, H+j, 2H+j, 3H+j of W), then does the cell math in-register.
__global__ void gates_cell_kernel(const unsigned short* __restrict__ x_bf,  // [B,E]
                                  const unsigned short* __restrict__ h_bf,  // [B,H]
                                  const unsigned short* __restrict__ Wih,   // [4H,E]
                                  const unsigned short* __restrict__ Whh,   // [4H,H]
                                  const float* __restrict__ b_ih,
                                  const float* __restrict__ b_hh,
                                  float* __restrict__ c,                    // [B,H] in/out
                                  unsigned short* __restrict__ hsht_bf,     // [B,1024]; h_t -> cols 512..1023
                                  float* __restrict__ cN)                   // last step only (else null)
{
  int wave = (int)((blockIdx.x * blockDim.x + threadIdx.x) >> 5);
  const int ntiles = H_ / 16;                 // 32
  if (wave >= (B_ / 16) * ntiles) return;
  int lane  = threadIdx.x & 31;
  int mt    = wave / ntiles;
  int jt    = wave % ntiles;
  int row   = lane & 15;
  int khalf = (lane >> 4) << 4;

  v8f acc0 = {}, acc1 = {}, acc2 = {}, acc3 = {};

  // x @ W_ih^T
  {
    const unsigned short* aRow = x_bf + (size_t)(mt * 16 + row) * E_ + khalf;
    const unsigned short* w0 = Wih + (size_t)(0 * H_ + jt * 16 + row) * E_ + khalf;
    const unsigned short* w1 = Wih + (size_t)(1 * H_ + jt * 16 + row) * E_ + khalf;
    const unsigned short* w2 = Wih + (size_t)(2 * H_ + jt * 16 + row) * E_ + khalf;
    const unsigned short* w3 = Wih + (size_t)(3 * H_ + jt * 16 + row) * E_ + khalf;
    for (int k = 0; k < E_; k += 32) {
      v16bf a = *reinterpret_cast<const v16bf*>(aRow + k);
      acc0 = __builtin_amdgcn_wmma_f32_16x16x32_bf16(false, a, false, *reinterpret_cast<const v16bf*>(w0 + k), (short)0, acc0, false, false);
      acc1 = __builtin_amdgcn_wmma_f32_16x16x32_bf16(false, a, false, *reinterpret_cast<const v16bf*>(w1 + k), (short)0, acc1, false, false);
      acc2 = __builtin_amdgcn_wmma_f32_16x16x32_bf16(false, a, false, *reinterpret_cast<const v16bf*>(w2 + k), (short)0, acc2, false, false);
      acc3 = __builtin_amdgcn_wmma_f32_16x16x32_bf16(false, a, false, *reinterpret_cast<const v16bf*>(w3 + k), (short)0, acc3, false, false);
    }
  }
  // h @ W_hh^T
  {
    const unsigned short* aRow = h_bf + (size_t)(mt * 16 + row) * H_ + khalf;
    const unsigned short* w0 = Whh + (size_t)(0 * H_ + jt * 16 + row) * H_ + khalf;
    const unsigned short* w1 = Whh + (size_t)(1 * H_ + jt * 16 + row) * H_ + khalf;
    const unsigned short* w2 = Whh + (size_t)(2 * H_ + jt * 16 + row) * H_ + khalf;
    const unsigned short* w3 = Whh + (size_t)(3 * H_ + jt * 16 + row) * H_ + khalf;
    for (int k = 0; k < H_; k += 32) {
      v16bf a = *reinterpret_cast<const v16bf*>(aRow + k);
      acc0 = __builtin_amdgcn_wmma_f32_16x16x32_bf16(false, a, false, *reinterpret_cast<const v16bf*>(w0 + k), (short)0, acc0, false, false);
      acc1 = __builtin_amdgcn_wmma_f32_16x16x32_bf16(false, a, false, *reinterpret_cast<const v16bf*>(w1 + k), (short)0, acc1, false, false);
      acc2 = __builtin_amdgcn_wmma_f32_16x16x32_bf16(false, a, false, *reinterpret_cast<const v16bf*>(w2 + k), (short)0, acc2, false, false);
      acc3 = __builtin_amdgcn_wmma_f32_16x16x32_bf16(false, a, false, *reinterpret_cast<const v16bf*>(w3 + k), (short)0, acc3, false, false);
    }
  }

  // LSTM cell (torch chunk order: i, f, g, o)
  int   n  = jt * 16 + (lane & 15);
  float bi = b_ih[0 * H_ + n] + b_hh[0 * H_ + n];
  float bf = b_ih[1 * H_ + n] + b_hh[1 * H_ + n];
  float bg = b_ih[2 * H_ + n] + b_hh[2 * H_ + n];
  float bo = b_ih[3 * H_ + n] + b_hh[3 * H_ + n];
  int mbase = mt * 16 + ((lane >> 4) << 3);
#pragma unroll
  for (int r = 0; r < 8; ++r) {
    int   m  = mbase + r;
    float iv = sigmoidf_(acc0[r] + bi);
    float fv = sigmoidf_(acc1[r] + bf);
    float gv = tanhf   (acc2[r] + bg);
    float ov = sigmoidf_(acc3[r] + bo);
    float cp = c[(size_t)m * H_ + n];
    float ct = fv * cp + iv * gv;
    float ht = tanhf(ct) * ov;
    c[(size_t)m * H_ + n] = ct;
    hsht_bf[(size_t)m * (2 * H_) + H_ + n] = f32_to_bf16_rne(ht);
    if (cN) cN[(size_t)m * H_ + n] = ct;
  }
}

// ---------- attention / softmax / pointer step ----------
// One block per batch row b; blockDim = 128 (== L).
__global__ void attn_kernel(const float* __restrict__ ctx,   // [B,L,H] projected context
                            const float* __restrict__ q,     // [B,H]
                            const float* __restrict__ Vv,    // [H]
                            float* __restrict__ mask,        // [B,L] (updated in place)
                            const float* __restrict__ emb,   // [B,L,E]
                            unsigned short* __restrict__ x_bf,    // [B,E] next input (bf16)
                            unsigned short* __restrict__ hsht_bf, // [B,1024]; hs -> cols 0..511
                            float* __restrict__ outAlphas,   // [B,L,L]
                            float* __restrict__ outPtrs,     // [B,L]
                            int t)
{
  __shared__ float qs[H_], vs[H_], sc[L_], red[L_];
  __shared__ int   redi[L_];
  int b = blockIdx.x, tid = threadIdx.x;

  for (int h = tid; h < H_; h += L_) { qs[h] = q[(size_t)b * H_ + h]; vs[h] = Vv[h]; }
  __syncthreads();

  // score for column l = tid
  const float* crow = ctx + ((size_t)b * L_ + tid) * H_;
  float s = 0.0f;
  for (int h = 0; h < H_; ++h) s += vs[h] * tanhf(qs[h] + crow[h]);
  float mk = mask[(size_t)b * L_ + tid];
  float sm = (mk > 0.0f) ? s : -__builtin_inff();

  // softmax: max
  red[tid] = sm;
  __syncthreads();
  for (int off = L_ / 2; off > 0; off >>= 1) {
    if (tid < off) red[tid] = fmaxf(red[tid], red[tid + off]);
    __syncthreads();
  }
  float mx = red[0];
  __syncthreads();
  float e = __expf(sm - mx);                 // masked lanes -> 0
  red[tid] = e;
  __syncthreads();
  for (int off = L_ / 2; off > 0; off >>= 1) {
    if (tid < off) red[tid] += red[tid + off];
    __syncthreads();
  }
  float alpha = e / red[0];
  __syncthreads();
  sc[tid] = alpha;
  outAlphas[((size_t)b * L_ + t) * L_ + tid] = alpha;

  // argmax(alpha * mask)
  red[tid] = alpha * mk; redi[tid] = tid;
  __syncthreads();
  for (int off = L_ / 2; off > 0; off >>= 1) {
    if (tid < off && red[tid + off] > red[tid]) { red[tid] = red[tid + off]; redi[tid] = redi[tid + off]; }
    __syncthreads();
  }
  int idx = redi[0];
  __syncthreads();

  // hs = sum_l alpha[l] * ctx[b,l,:]
  for (int h = tid; h < H_; h += L_) {
    float acc = 0.0f;
    for (int l = 0; l < L_; ++l) acc += sc[l] * ctx[((size_t)b * L_ + l) * H_ + h];
    hsht_bf[(size_t)b * (2 * H_) + h] = f32_to_bf16_rne(acc);
  }
  // next input x = embedded_inputs[b, idx, :]
  const float* er = emb + ((size_t)b * L_ + idx) * E_;
  for (int ee = tid; ee < E_; ee += L_) x_bf[(size_t)b * E_ + ee] = f32_to_bf16_rne(er[ee]);

  if (tid == 0) {
    outPtrs[(size_t)b * L_ + t] = (float)idx;
    mask[(size_t)b * L_ + idx] = 0.0f;
  }
}

// ---------- host ----------
extern "C" void kernel_launch(void* const* d_in, const int* in_sizes, int n_in,
                              void* d_out, int out_size, void* d_ws, size_t ws_size,
                              hipStream_t stream) {
  (void)in_sizes; (void)n_in; (void)out_size; (void)ws_size;
  const float* emb    = (const float*)d_in[0];   // [B,L,E]
  const float* dec    = (const float*)d_in[1];   // [B,E]
  const float* h0     = (const float*)d_in[2];   // [B,H]
  const float* c0     = (const float*)d_in[3];   // [B,H]
  const float* contex = (const float*)d_in[4];   // [B,L,H]
  const float* W_ih   = (const float*)d_in[5];
  const float* b_ih   = (const float*)d_in[6];
  const float* W_hh   = (const float*)d_in[7];
  const float* b_hh   = (const float*)d_in[8];
  const float* W_ho   = (const float*)d_in[9];
  const float* b_ho   = (const float*)d_in[10];
  const float* W_in   = (const float*)d_in[11];
  const float* b_in   = (const float*)d_in[12];
  const float* W_ctx  = (const float*)d_in[13];
  const float* b_ctx  = (const float*)d_in[14];
  const float* Vv     = (const float*)d_in[15];

  // d_out layout: outputs [B,L,L], pointers [B,L], hN [B,H], cN [B,H]
  float* outAlphas = (float*)d_out;
  float* outPtrs   = outAlphas + (size_t)B_ * L_ * L_;
  float* outHN     = outPtrs + (size_t)B_ * L_;
  float* outCN     = outHN + (size_t)B_ * H_;

  // workspace carve-up (256B aligned)
  char* ws = (char*)d_ws;
  size_t off = 0;
  auto alloc = [&](size_t bytes) -> void* {
    void* p = ws + off; off += (bytes + 255) & ~(size_t)255; return p;
  };
  float*          ctxf     = (float*)alloc((size_t)B_ * L_ * H_ * 4);          // projected ctx
  unsigned short* ctxin_bf = (unsigned short*)alloc((size_t)B_ * L_ * H_ * 2); // context in bf16
  unsigned short* Wih_bf   = (unsigned short*)alloc((size_t)4 * H_ * E_ * 2);
  unsigned short* Whh_bf   = (unsigned short*)alloc((size_t)4 * H_ * H_ * 2);
  unsigned short* Win_bf   = (unsigned short*)alloc((size_t)H_ * H_ * 2);
  unsigned short* Wctx_bf  = (unsigned short*)alloc((size_t)H_ * H_ * 2);
  unsigned short* Who_bf   = (unsigned short*)alloc((size_t)H_ * 2 * H_ * 2);
  unsigned short* x_bf     = (unsigned short*)alloc((size_t)B_ * E_ * 2);
  unsigned short* h_bf     = (unsigned short*)alloc((size_t)B_ * H_ * 2);
  unsigned short* hsht_bf  = (unsigned short*)alloc((size_t)B_ * 2 * H_ * 2);  // [hs | h_t]
  float*          c_st     = (float*)alloc((size_t)B_ * H_ * 4);
  float*          q_st     = (float*)alloc((size_t)B_ * H_ * 4);
  float*          mask     = (float*)alloc((size_t)B_ * L_ * 4);

  auto cvt = [&](const float* s, unsigned short* d, int n) {
    convert_f32_bf16<<<(n + 255) / 256, 256, 0, stream>>>(s, d, n);
  };
  cvt(contex, ctxin_bf, B_ * L_ * H_);
  cvt(W_ih,  Wih_bf,  4 * H_ * E_);
  cvt(W_hh,  Whh_bf,  4 * H_ * H_);
  cvt(W_in,  Win_bf,  H_ * H_);
  cvt(W_ctx, Wctx_bf, H_ * H_);
  cvt(W_ho,  Who_bf,  H_ * 2 * H_);
  cvt(h0,    h_bf,    B_ * H_);
  cvt(dec,   x_bf,    B_ * E_);
  init_state<<<(B_ * H_ + 255) / 256, 256, 0, stream>>>(c0, c_st, mask);

  // ctx = context @ W_ctx^T + b_ctx : M=B*L=16384, N=H=512, K=H=512
  {
    int mtGroups = (B_ * L_) / 16 / 4;          // 256 (4 waves per block in M)
    int ntiles   = H_ / 16;                     // 32
    wmma_gemm_lds<0, H_><<<mtGroups * ntiles, 128, 0, stream>>>(
        ctxin_bf, H_, Wctx_bf, b_ctx, ntiles, nullptr, 0, ctxf, H_);
  }

  const int gemmTiles  = (B_ / 16) * (H_ / 16); // 256 waves for gates kernel
  const int stepBlocks = ((B_ / 16) / 4) * (H_ / 16); // 2 * 32 = 64 blocks
  for (int t = 0; t < L_; ++t) {
    // LSTM gates + cell: writes c_st, h_t half of hsht_bf (and cN at last step)
    gates_cell_kernel<<<gemmTiles / 4, 128, 0, stream>>>(
        x_bf, h_bf, Wih_bf, Whh_bf, b_ih, b_hh, c_st, hsht_bf,
        (t == L_ - 1) ? outCN : nullptr);

    // q = h_t @ W_in^T + b_in   (A = h_t half of concat buffer, lda = 2H)
    wmma_gemm_lds<0, H_><<<stepBlocks, 128, 0, stream>>>(
        hsht_bf + H_, 2 * H_, Win_bf, b_in, H_ / 16, nullptr, 0, q_st, H_);

    // attention: alpha -> d_out, hs -> concat buffer, ptr/mask/x update
    attn_kernel<<<B_, L_, 0, stream>>>(ctxf, q_st, Vv, mask, emb, x_bf,
                                       hsht_bf, outAlphas, outPtrs, t);

    // h_new = tanh([hs|h_t] @ W_ho^T + b_ho) -> h_bf (and hN f32 at last step)
    wmma_gemm_lds<1, 2 * H_><<<stepBlocks, 128, 0, stream>>>(
        hsht_bf, 2 * H_, Who_bf, b_ho, H_ / 16, h_bf, H_,
        (t == L_ - 1) ? outHN : nullptr, H_);
  }
}
